// SelfAttention_103079215345
// MI455X (gfx1250) — matrix operand order repaired and checked by reading, not compiled, not run on hardware
//
#include <hip/hip_runtime.h>

// ---------------------------------------------------------------------------
// CDNA5 (gfx1250) bf16 WMMA self-attention forward.
//   - all matmuls: v_wmma_f32_16x16x32_bf16 (fp32 accumulate)
//   - GEMMs: TDM (tensor_load_to_lds) double-buffered LDS staging
//   - attention: per-wave flash attention, LDS-staged P, exp2-domain softmax
// ---------------------------------------------------------------------------

typedef __attribute__((ext_vector_type(16))) __bf16 v16bf;
typedef __attribute__((ext_vector_type(8)))  __bf16 v8bf;
typedef __attribute__((ext_vector_type(4)))  __bf16 v4bf;
typedef __attribute__((ext_vector_type(8)))  float  v8f;
typedef __attribute__((ext_vector_type(4)))  unsigned int v4u;
typedef __attribute__((ext_vector_type(8)))  int    v8i;
typedef __attribute__((ext_vector_type(4)))  int    v4i;

union frag_u { v16bf v; v8bf h[2]; };

// fp32 -> bf16, round-to-nearest-even via bit math
__device__ inline __bf16 f2bf(float f) {
    union { float f; unsigned u; } uf; uf.f = f;
    unsigned r = uf.u + 0x7FFFu + ((uf.u >> 16) & 1u);
    union { unsigned short s; __bf16 b; } ub;
    ub.s = (unsigned short)(r >> 16);
    return ub.b;
}

// A fragment (16x32 bf16, MxK) from row-major src[row][k], ld in elements.
// Lane half h, row r=lane&15 holds K = {h*8..h*8+7} U {16+h*8..16+h*8+7}.
__device__ inline v16bf load_a_frag(const __bf16* src, int ld, int row0, int k0, int lane) {
    const int h = (lane >> 4) & 1;
    const int r = lane & 15;
    const __bf16* p = src + (size_t)(row0 + r) * ld + k0;
    frag_u f;
    f.h[0] = *reinterpret_cast<const v8bf*>(p + h * 8);
    f.h[1] = *reinterpret_cast<const v8bf*>(p + 16 + h * 8);
    return f.v;
}

// B fragment (32x16 bf16, KxN) where B[k][n] = src[n][k] (src row-major).
// Lane half h, col n=lane&15 holds K = {h*8..h*8+7} U {16+h*8..16+h*8+7}.
__device__ inline v16bf load_b_frag(const __bf16* src, int ld, int n0, int k0, int lane) {
    const int h = (lane >> 4) & 1;
    const int n = lane & 15;
    const __bf16* p = src + (size_t)(n0 + n) * ld + k0;
    frag_u f;
    f.h[0] = *reinterpret_cast<const v8bf*>(p + h * 8);
    f.h[1] = *reinterpret_cast<const v8bf*>(p + 16 + h * 8);
    return f.v;
}

__device__ inline v8f wmma_bf16(v16bf a, v16bf b, v8f c) {
    return __builtin_amdgcn_wmma_f32_16x16x32_bf16(
        false, a, false, b, (short)0, c, false, false);
}

// low 32 bits of a generic pointer to __shared__ == LDS byte offset
__device__ inline unsigned lds_off(const void* p) {
    return (unsigned)(unsigned long long)p;
}

// 2D TDM load: tile_d1 rows x tile_d0 elems (bf16), row stride in elements.
// Builds D# group0/group1 per CDNA5 ISA.
__device__ inline void tdm_load_2d(unsigned lds_byte_off, const void* gptr,
                                   unsigned tile_d0, unsigned tile_d1,
                                   unsigned long long row_stride) {
    unsigned long long ga = (unsigned long long)gptr;
    v4u g0;
    g0[0] = 1u;                                   // count=1, user mode
    g0[1] = lds_byte_off;                         // lds_addr
    g0[2] = (unsigned)ga;                         // global_addr[31:0]
    g0[3] = (unsigned)((ga >> 32) & 0x1FFFFFFu)   // global_addr[56:32]
          | (2u << 30);                           // type = 2 ("image")
    v8i g1;
    g1[0] = (int)(1u << 16);                      // data_size = 1 (2 bytes)
    g1[1] = (int)(tile_d0 << 16);                 // tensor_dim0[15:0]  = tile_d0
    g1[2] = (int)(tile_d1 << 16);                 // tensor_dim1[15:0]  = tile_d1
    g1[3] = (int)(tile_d0 << 16);                 // tile_dim0
    g1[4] = (int)(tile_d1 & 0xFFFFu);             // tile_dim1 (tile_dim2 = 0)
    g1[5] = (int)(unsigned)(row_stride);          // tensor_dim0_stride[31:0]
    g1[6] = (int)(unsigned)((row_stride >> 32) & 0xFFFFu);
    g1[7] = 0;
    v4i z4 = {0, 0, 0, 0};
#if defined(__clang_major__) && (__clang_major__ >= 23)
    v8i z8 = {0, 0, 0, 0, 0, 0, 0, 0};
    __builtin_amdgcn_tensor_load_to_lds(g0, g1, z4, z4, z8, 0);
#else
    __builtin_amdgcn_tensor_load_to_lds(g0, g1, z4, z4, 0);
#endif
}

// ---------------------------------------------------------------------------
// fp32 -> bf16 conversion, 4 elements/thread
// ---------------------------------------------------------------------------
__global__ __launch_bounds__(256) void cvt_f32_bf16(const float* __restrict__ in,
                                                    __bf16* __restrict__ out, int n) {
    int i = (blockIdx.x * blockDim.x + threadIdx.x) * 4;
    if (i + 3 < n) {
        float4 f = *reinterpret_cast<const float4*>(in + i);
        v4bf o;
        o[0] = f2bf(f.x); o[1] = f2bf(f.y); o[2] = f2bf(f.z); o[3] = f2bf(f.w);
        *reinterpret_cast<v4bf*>(out + i) = o;
    }
}

// ---------------------------------------------------------------------------
// TDM-fed GEMM: C[8192,1024] = X @ W^T (K=1024), bf16 in, f32 acc.
// Block = 8 waves; block tile 256x64; wave tile 32x64 (8 accumulators).
// Double-buffered LDS (X 256x32, W 64x32); wave 0 drives the TDM.
// Epilogue: mode 0 -> bf16 [B*H,S,64]; mode 1 -> bf16 [B*H,64,S];
//           mode 2 -> fp32 [8192,1024] + bias.
// ---------------------------------------------------------------------------
__global__ __launch_bounds__(256) void gemm_tdm(const __bf16* __restrict__ X,
                                                const __bf16* __restrict__ W,
                                                __bf16* __restrict__ outb,
                                                float* __restrict__ outf,
                                                const float* __restrict__ bias,
                                                int mode) {
    __shared__ __bf16 Xt[2][256 * 32];   // 2 x 16 KB
    __shared__ __bf16 Wt[2][64 * 32];    // 2 x  4 KB
    const int lane = threadIdx.x & 31;
    const int wave = threadIdx.x >> 5;
    const int row0 = blockIdx.y * 256;
    const int col0 = blockIdx.x * 64;

    if (wave == 0) {                     // prologue: fill buffer 0
        tdm_load_2d(lds_off(&Xt[0][0]), X + (size_t)row0 * 1024, 32, 256, 1024);
        tdm_load_2d(lds_off(&Wt[0][0]), W + (size_t)col0 * 1024, 32, 64, 1024);
        __builtin_amdgcn_s_wait_tensorcnt(0);
    }
    __syncthreads();

    v8f acc[2][4] = {};
    for (int kk = 0; kk < 32; kk++) {    // 32 K-chunks of 32
        const int buf = kk & 1;
        if (wave == 0 && kk + 1 < 32) {  // prefetch next chunk via TDM
            const int k0n = (kk + 1) * 32;
            tdm_load_2d(lds_off(&Xt[buf ^ 1][0]), X + (size_t)row0 * 1024 + k0n, 32, 256, 1024);
            tdm_load_2d(lds_off(&Wt[buf ^ 1][0]), W + (size_t)col0 * 1024 + k0n, 32, 64, 1024);
        }
        const __bf16* xs = &Xt[buf][0];
        const __bf16* ws = &Wt[buf][0];
        v16bf a0 = load_a_frag(xs, 32, wave * 32,      0, lane);
        v16bf a1 = load_a_frag(xs, 32, wave * 32 + 16, 0, lane);
        v16bf b0 = load_b_frag(ws, 32,  0, 0, lane);
        v16bf b1 = load_b_frag(ws, 32, 16, 0, lane);
        v16bf b2 = load_b_frag(ws, 32, 32, 0, lane);
        v16bf b3 = load_b_frag(ws, 32, 48, 0, lane);
        acc[0][0] = wmma_bf16(a0, b0, acc[0][0]);
        acc[0][1] = wmma_bf16(a0, b1, acc[0][1]);
        acc[0][2] = wmma_bf16(a0, b2, acc[0][2]);
        acc[0][3] = wmma_bf16(a0, b3, acc[0][3]);
        acc[1][0] = wmma_bf16(a1, b0, acc[1][0]);
        acc[1][1] = wmma_bf16(a1, b1, acc[1][1]);
        acc[1][2] = wmma_bf16(a1, b2, acc[1][2]);
        acc[1][3] = wmma_bf16(a1, b3, acc[1][3]);
        if (wave == 0 && kk + 1 < 32)
            __builtin_amdgcn_s_wait_tensorcnt(0);    // next buffer landed
        __syncthreads();                             // implies dscnt wait: WAR-safe
    }

    const int h = (lane >> 4) & 1;
    const int c = lane & 15;
    for (int i = 0; i < 2; i++)
        for (int j = 0; j < 4; j++)
            for (int v = 0; v < 8; v++) {
                int m = row0 + wave * 32 + i * 16 + v + 8 * h;   // b*2048+s
                int n = col0 + j * 16 + c;                       // head*64+d
                if (mode == 2) {
                    outf[(size_t)m * 1024 + n] = acc[i][j][v] + bias[n];
                } else {
                    int b = m >> 11, s = m & 2047;
                    int hd = n >> 6,  d = n & 63;
                    size_t idx = (mode == 0)
                        ? ((size_t)(b * 16 + hd) * 2048 + s) * 64 + d
                        : ((size_t)(b * 16 + hd) * 64 + d) * 2048 + s;
                    outb[idx] = f2bf(acc[i][j][v]);
                }
            }
}

// ---------------------------------------------------------------------------
// Flash attention (causal). One wave per 16-row query tile; 32 keys / iter.
// Q,K: [BH, 2048, 64] bf16   Vt: [BH, 64, 2048] bf16   O: [B*S, 1024] bf16
// Softmax in exp2 domain (v_exp_f32 native).
// ---------------------------------------------------------------------------
__global__ __launch_bounds__(128) void attn_fwd(const __bf16* __restrict__ Q,
                                                const __bf16* __restrict__ K,
                                                const __bf16* __restrict__ Vt,
                                                __bf16* __restrict__ O) {
    const int lane = threadIdx.x & 31;
    const int wave = threadIdx.x >> 5;
    const int bh   = blockIdx.y;
    const int q0   = blockIdx.x * 64 + wave * 16;

    const __bf16* Qh = Q  + (size_t)bh * 2048 * 64;
    const __bf16* Kh = K  + (size_t)bh * 2048 * 64;
    const __bf16* Vh = Vt + (size_t)bh * 64 * 2048;

    __shared__ __bf16 pstage[4][16 * 32];
    __bf16* pst = pstage[wave];

    v16bf qa0 = load_a_frag(Qh, 64, q0, 0,  lane);
    v16bf qa1 = load_a_frag(Qh, 64, q0, 32, lane);

    v8f acc[4] = {};
    float mrow[8], lrow[8];
    for (int v = 0; v < 8; v++) { mrow[v] = -3.0e38f; lrow[v] = 0.0f; }

    const int h = (lane >> 4) & 1;
    const int c = lane & 15;
    const int qlast = q0 + 15;
    const float SC = 0.03125f * 1.44269504088896340736f;  // (1/sqrt(1024))*log2(e)

    for (int j0 = 0; j0 <= qlast; j0 += 32) {
        v8f sc[2];
        for (int t = 0; t < 2; t++) {
            v16bf kb0 = load_b_frag(Kh, 64, j0 + t * 16, 0,  lane);
            v16bf kb1 = load_b_frag(Kh, 64, j0 + t * 16, 32, lane);
            v8f s = {};
            s = wmma_bf16(qa0, kb0, s);
            s = wmma_bf16(qa1, kb1, s);
            sc[t] = s;
        }
        for (int v = 0; v < 8; v++) {
            int qrow = q0 + v + 8 * h;
            float s0 = sc[0][v] * SC;
            float s1 = sc[1][v] * SC;
            if (j0 + c      > qrow) s0 = -3.0e38f;
            if (j0 + 16 + c > qrow) s1 = -3.0e38f;
            float x = fmaxf(s0, s1);
            x = fmaxf(x, __shfl_xor(x, 1, 32));
            x = fmaxf(x, __shfl_xor(x, 2, 32));
            x = fmaxf(x, __shfl_xor(x, 4, 32));
            x = fmaxf(x, __shfl_xor(x, 8, 32));
            float mnew = fmaxf(mrow[v], x);
            float corr = __builtin_exp2f(mrow[v] - mnew);
            float p0 = __builtin_exp2f(s0 - mnew);
            float p1 = __builtin_exp2f(s1 - mnew);
            float rs = p0 + p1;
            rs += __shfl_xor(rs, 1, 32);
            rs += __shfl_xor(rs, 2, 32);
            rs += __shfl_xor(rs, 4, 32);
            rs += __shfl_xor(rs, 8, 32);
            lrow[v] = lrow[v] * corr + rs;
            mrow[v] = mnew;
            acc[0][v] *= corr; acc[1][v] *= corr;
            acc[2][v] *= corr; acc[3][v] *= corr;
            int prow = v + 8 * h;
            pst[prow * 32 + c]      = f2bf(p0);
            pst[prow * 32 + 16 + c] = f2bf(p1);
        }
        asm volatile("s_wait_dscnt 0" ::: "memory");   // same-wave LDS RAW
        v16bf pa = load_a_frag(pst, 32, 0, 0, lane);
        for (int t2 = 0; t2 < 4; t2++) {
            v16bf vb = load_b_frag(Vh, 2048, t2 * 16, j0, lane);
            acc[t2] = wmma_bf16(pa, vb, acc[t2]);
        }
    }

    const int b = bh >> 4, head = bh & 15;
    for (int t2 = 0; t2 < 4; t2++)
        for (int v = 0; v < 8; v++) {
            int row = q0 + v + 8 * h;
            float o = acc[t2][v] / lrow[v];
            size_t idx = (size_t)(b * 2048 + row) * 1024 + head * 64 + t2 * 16 + c;
            O[idx] = f2bf(o);
        }
}

// ---------------------------------------------------------------------------
extern "C" void kernel_launch(void* const* d_in, const int* in_sizes, int n_in,
                              void* d_out, int out_size, void* d_ws, size_t ws_size,
                              hipStream_t stream) {
    const float* x  = (const float*)d_in[0];
    const float* Wq = (const float*)d_in[1];
    const float* Wk = (const float*)d_in[2];
    const float* Wv = (const float*)d_in[3];
    const float* Wo = (const float*)d_in[4];
    const float* bo = (const float*)d_in[5];
    float* out = (float*)d_out;

    char* ws = (char*)d_ws;
    const size_t MB = 1024 * 1024;
    __bf16* xb  = (__bf16*)(ws);            // 16 MB  x bf16 [8192,1024]
    __bf16* wqb = (__bf16*)(ws + 16 * MB);
    __bf16* wkb = (__bf16*)(ws + 18 * MB);
    __bf16* wvb = (__bf16*)(ws + 20 * MB);
    __bf16* wob = (__bf16*)(ws + 22 * MB);
    __bf16* Qb  = (__bf16*)(ws + 24 * MB);  // [BH,S,64]
    __bf16* Kb  = (__bf16*)(ws + 40 * MB);  // [BH,S,64]
    __bf16* Vtb = (__bf16*)(ws + 56 * MB);  // [BH,64,S]
    __bf16* Ob  = (__bf16*)(ws + 72 * MB);  // [B*S,1024]  (88 MB total)

    const int nx = 4 * 2048 * 1024;
    const int nw = 1024 * 1024;
    cvt_f32_bf16<<<nx / 4 / 256, 256, 0, stream>>>(x,  xb,  nx);
    cvt_f32_bf16<<<nw / 4 / 256, 256, 0, stream>>>(Wq, wqb, nw);
    cvt_f32_bf16<<<nw / 4 / 256, 256, 0, stream>>>(Wk, wkb, nw);
    cvt_f32_bf16<<<nw / 4 / 256, 256, 0, stream>>>(Wv, wvb, nw);
    cvt_f32_bf16<<<nw / 4 / 256, 256, 0, stream>>>(Wo, wob, nw);

    dim3 gg(16, 32);                        // N/64 x M/256
    gemm_tdm<<<gg, 256, 0, stream>>>(xb, wqb, Qb,  nullptr, nullptr, 0);
    gemm_tdm<<<gg, 256, 0, stream>>>(xb, wkb, Kb,  nullptr, nullptr, 0);
    gemm_tdm<<<gg, 256, 0, stream>>>(xb, wvb, Vtb, nullptr, nullptr, 1);

    attn_fwd<<<dim3(32, 64), 128, 0, stream>>>(Qb, Kb, Vtb, Ob);

    gemm_tdm<<<gg, 256, 0, stream>>>(Ob, wob, nullptr, out, bo, 2);
}
